// BCNN_64845416235415
// MI455X (gfx1250) — matrix operand order repaired and checked by reading, not compiled, not run on hardware
//
#include <hip/hip_runtime.h>
#include <hip/hip_bf16.h>

// ---------------------------------------------------------------------------
// Binarized CNN forward pass for MI455X (gfx1250, wave32, WMMA).
//
// Fuse maxpool+BN+sign into producers; run the first conv (im2col, +/-1
// weights) and all three binarized 1x1 convs as v_wmma_f32_16x16x32_f16
// GEMMs (operands exactly representable in f16, f32 accumulation). Whole net
// ~2.3 GFLOP / <100 MB traffic: bandwidth/latency bound, fits in 192 MB L2.
// ---------------------------------------------------------------------------

typedef __attribute__((ext_vector_type(16))) _Float16 v16h;
typedef __attribute__((ext_vector_type(8)))  _Float16 v8h;
typedef __attribute__((ext_vector_type(8)))  float    v8f;

union VH16 { v16h v; v8h h8[2]; _Float16 h[16]; };

#define BN_EPS 1e-3f

#define BATCH 64
#define X_H   96
#define X_W   192
#define P1_H  47
#define P1_W  95

// ---------------------------------------------------------------------------
// Pack sign(W) [Ksrc x N] f32 -> WMMA B-fragment layout, K zero-padded to
// ktiles*32.  Fragment f = kt*ntiles+nt; lane L holds column n = nt*16+(L&15),
// elements e -> K = kt*32 + (e<8 ? 8*(L>>4)+e : 16+8*(L>>4)+e-8)   (dense 16-bit B 32x16 layout).
// ---------------------------------------------------------------------------
__global__ void pack_signB(const float* __restrict__ w, _Float16* __restrict__ dst,
                           int Ksrc, int N, int ktiles, int ntiles) {
    int t = blockIdx.x * blockDim.x + threadIdx.x;
    int total = ktiles * ntiles * 32;
    if (t >= total) return;
    int lane = t & 31;
    int frag = t >> 5;
    int nt = frag % ntiles;
    int kt = frag / ntiles;
    int hs = lane >> 4;
    int n  = nt * 16 + (lane & 15);
    _Float16* out = dst + (size_t)t * 16;
    for (int e = 0; e < 16; ++e) {
        int k = kt * 32 + (e < 8 ? 8 * hs + e : 16 + 8 * hs + (e - 8));
        _Float16 v = (_Float16)0.0f;
        if (k < Ksrc)
            v = (w[(size_t)k * N + n] >= 0.0f) ? (_Float16)1.0f : (_Float16)-1.0f;
        out[e] = v;
    }
}

// ---------------------------------------------------------------------------
// First conv (3x3x3 -> 32ch, VALID, sign weights) + 2x2 maxpool + BN, fused.
// One wave per (b, pooled row, 16-wide pooled-col tile). 64 conv positions
// (2 rows x 32 cols) = 4 WMMA M-tiles; N=32 = 2 N-tiles; K=27 padded to 32.
// Input patch staged in LDS as f16 (convert once); A-fragment gather is
// branch-free: per-element compile-time offsets for both lane halves, one
// cndmask-selected ds_load_u16 each, value-masked K padding.
// ---------------------------------------------------------------------------
__global__ void __launch_bounds__(32)
conv1_pool_bn(const float* __restrict__ x, const _Float16* __restrict__ w1p,
              const float* __restrict__ beta, const float* __restrict__ mean,
              const float* __restrict__ var, float* __restrict__ out) {
    __shared__ _Float16 lx[4][34][3];   // input patch (f16): 4 rows x 34 cols x 3 ch
    __shared__ float    lc[64][32];     // conv outputs before pooling
    const int tw   = blockIdx.x;        // pooled-col tile (0..5)
    const int ph   = blockIdx.y;        // pooled row (0..46)
    const int b    = blockIdx.z;        // batch
    const int lane = threadIdx.x;
    const int pw0  = tw * 16;
    const int r0   = 2 * ph;
    const int c0   = 2 * pw0;

    // cooperative stage of the input patch into LDS (clamp right edge)
    for (int i = lane; i < 4 * 34 * 3; i += 32) {
        int ci = i % 3;
        int cc = (i / 3) % 34;
        int rr = i / (3 * 34);
        int gc = c0 + cc; if (gc > X_W - 1) gc = X_W - 1;
        lx[rr][cc][ci] =
            (_Float16)x[(((size_t)(b * X_H + (r0 + rr))) * X_W + gc) * 3 + ci];
    }
    __syncthreads();

    // B fragments (packed sign weights), 32 bytes contiguous per lane
    v16h bw0 = *(const v16h*)(w1p + (size_t)(0 * 32 + lane) * 16);
    v16h bw1 = *(const v16h*)(w1p + (size_t)(1 * 32 + lane) * 16);

    const int hs = lane >> 4;
    const int ml = lane & 15;
    const _Float16* lxf = &lx[0][0][0];

    for (int mt = 0; mt < 4; ++mt) {
        int q  = mt * 16 + ml;          // conv position within the 64-pos tile
        int rr = q >> 5;                // conv row (0/1)
        int cc = q & 31;                // conv col within tile
        const _Float16* base = lxf + rr * 102 + cc * 3;  // &lx[rr][cc][0]
        VH16 a;                         // 16-bit A 16x32 fragment
        #pragma unroll
        for (int e = 0; e < 16; ++e) {
            // K for the two lane halves; all offsets fold to literals
            const int k0  = (e < 8 ? e : e + 8);          // hs = 0 -> K in 0..23
            const int k1  = k0 + 8;                       // hs = 1 -> K in 8..31
            const int k1c = (k1 < 27) ? k1 : 0;           // clamp padded K
            const int off0 = 102 * (k0 / 9) + (k0 % 9);   // kh*rowpitch + kw*3+ci
            const int off1 = 102 * (k1c / 9) + (k1c % 9);
            _Float16 f = base[hs ? off1 : off0];          // unconditional ds load
            a.h[e] = (hs && (k1 >= 27)) ? (_Float16)0.0f : f;  // mask K padding
        }
        v8f acc0 = {};
        v8f acc1 = {};
        acc0 = __builtin_amdgcn_wmma_f32_16x16x32_f16(false, a.v, false, bw0,
                                                      (short)0, acc0, false, false);
        acc1 = __builtin_amdgcn_wmma_f32_16x16x32_f16(false, a.v, false, bw1,
                                                      (short)0, acc1, false, false);
        #pragma unroll
        for (int v = 0; v < 8; ++v) {
            int mrow = mt * 16 + v + 8 * hs;   // C layout: lanes 16-31 hold M=v+8
            lc[mrow][ml]      = acc0[v];       // n = ml
            lc[mrow][16 + ml] = acc1[v];       // n = 16+ml
        }
    }
    __syncthreads();

    // 2x2 maxpool + BN; lane = output channel
    const int n = lane;
    const float mn = mean[n];
    const float rs = rsqrtf(var[n] + BN_EPS);
    const float bt = beta[n];
    #pragma unroll
    for (int pwl = 0; pwl < 16; ++pwl) {
        int pw = pw0 + pwl;
        float m0 = lc[2 * pwl][n],      m1 = lc[2 * pwl + 1][n];
        float m2 = lc[32 + 2 * pwl][n], m3 = lc[33 + 2 * pwl][n];
        float m = fmaxf(fmaxf(m0, m1), fmaxf(m2, m3));
        if (pw < P1_W)
            out[((size_t)(b * P1_H + ph) * P1_W + pw) * 32 + n] = (m - mn) * rs + bt;
    }
}

// ---------------------------------------------------------------------------
// Fused depthwise3x3(SAME)+bias -> 2x2 maxpool -> BN -> ste_sign, producing
// the +/-1 f16 A-operand tensor [M = B*H2*W2, K = C] for the pointwise GEMM.
// ---------------------------------------------------------------------------
__global__ void dw_pool_bn_sign(const float* __restrict__ in, const float* __restrict__ kern,
                                const float* __restrict__ bias,
                                const float* __restrict__ beta, const float* __restrict__ mean,
                                const float* __restrict__ var,
                                _Float16* __restrict__ out,
                                int H, int W, int C) {
    const int H2 = H >> 1, W2 = W >> 1;
    const int total = BATCH * H2 * W2 * C;
    int idx = blockIdx.x * blockDim.x + threadIdx.x;
    if (idx >= total) return;
    int c  = idx % C;
    int pw = (idx / C) % W2;
    int ph = (idx / (C * W2)) % H2;
    int b  = idx / (C * W2 * H2);

    // hint the streaming window into cache (global_prefetch_b8)
    __builtin_prefetch(&in[(((size_t)b * H + 2 * ph) * W + 2 * pw) * C + c], 0, 1);

    float kk[9];
    #pragma unroll
    for (int t = 0; t < 9; ++t) kk[t] = kern[t * C + c];
    const float bs = bias[c];

    float best = -3.0e38f;
    #pragma unroll
    for (int dy = 0; dy < 2; ++dy) {
        #pragma unroll
        for (int dx = 0; dx < 2; ++dx) {
            int y = 2 * ph + dy, x0 = 2 * pw + dx;
            float acc = bs;
            #pragma unroll
            for (int kh = -1; kh <= 1; ++kh) {
                #pragma unroll
                for (int kw = -1; kw <= 1; ++kw) {
                    int iy = y + kh, ix = x0 + kw;
                    if (iy >= 0 && iy < H && ix >= 0 && ix < W)
                        acc += in[(((size_t)b * H + iy) * W + ix) * C + c]
                               * kk[(kh + 1) * 3 + (kw + 1)];
                }
            }
            best = fmaxf(best, acc);
        }
    }
    float v = (best - mean[c]) * rsqrtf(var[c] + BN_EPS) + beta[c];
    out[idx] = (v >= 0.0f) ? (_Float16)1.0f : (_Float16)-1.0f;
}

// ---------------------------------------------------------------------------
// Binarized 1x1 conv as WMMA GEMM: out[M,N] = A[M,K] * signB[K,N], fused BN.
// One wave per 16x16 tile; K = KTILES*32 handled as statically chained
// 16x16x32 f16 WMMAs. A-fragment = two aligned 16B loads per lane.
// ---------------------------------------------------------------------------
template<int KTILES>
__global__ void __launch_bounds__(32)
pw_gemm_bn(const _Float16* __restrict__ A, const _Float16* __restrict__ Bp,
           const float* __restrict__ beta, const float* __restrict__ mean,
           const float* __restrict__ var, float* __restrict__ out,
           int N, int ntiles) {
    const int K    = KTILES * 32;
    const int mt   = blockIdx.x;
    const int nt   = blockIdx.y;
    const int lane = threadIdx.x;
    const int hs   = lane >> 4;
    const int ml   = lane & 15;
    const int m    = mt * 16 + ml;
    const _Float16* arow = A + (size_t)m * K;

    v8f acc = {};
    #pragma unroll
    for (int kt = 0; kt < KTILES; ++kt) {
        VH16 a;
        a.h8[0] = *(const v8h*)(arow + kt * 32 + 8 * hs);        // K = kt*32+8*hs+0..7
        a.h8[1] = *(const v8h*)(arow + kt * 32 + 16 + 8 * hs);   // K = kt*32+16+8*hs+0..7
        v16h bf = *(const v16h*)(Bp + (size_t)((kt * ntiles + nt) * 32 + lane) * 16);
        acc = __builtin_amdgcn_wmma_f32_16x16x32_f16(false, a.v, false, bf,
                                                     (short)0, acc, false, false);
    }

    const int n = nt * 16 + ml;
    const float mn = mean[n];
    const float rs = rsqrtf(var[n] + BN_EPS);
    const float bt = beta[n];
    #pragma unroll
    for (int v = 0; v < 8; ++v) {
        int mrow = mt * 16 + v + 8 * hs;
        out[(size_t)mrow * N + n] = (acc[v] - mn) * rs + bt;
    }
}

// ---------------------------------------------------------------------------
extern "C" void kernel_launch(void* const* d_in, const int* in_sizes, int n_in,
                              void* d_out, int out_size, void* d_ws, size_t ws_size,
                              hipStream_t stream) {
    (void)in_sizes; (void)n_in; (void)out_size; (void)ws_size;

    const float* x        = (const float*)d_in[0];
    const float* w1       = (const float*)d_in[1];
    const float* bn1_bt   = (const float*)d_in[2];
    const float* bn1_mn   = (const float*)d_in[3];
    const float* bn1_vr   = (const float*)d_in[4];
    // per block i: dw_k, dw_b, bnA_beta, bnA_mean, bnA_var, pw_k, bnB_beta, bnB_mean, bnB_var
    const float* dwk[3], *dwb[3], *Abt[3], *Amn[3], *Avr[3], *pwk[3], *Bbt[3], *Bmn[3], *Bvr[3];
    for (int i = 0; i < 3; ++i) {
        int base = 5 + 9 * i;
        dwk[i] = (const float*)d_in[base + 0];
        dwb[i] = (const float*)d_in[base + 1];
        Abt[i] = (const float*)d_in[base + 2];
        Amn[i] = (const float*)d_in[base + 3];
        Avr[i] = (const float*)d_in[base + 4];
        pwk[i] = (const float*)d_in[base + 5];
        Bbt[i] = (const float*)d_in[base + 6];
        Bmn[i] = (const float*)d_in[base + 7];
        Bvr[i] = (const float*)d_in[base + 8];
    }
    float* outp = (float*)d_out;

    // workspace carve-up
    char* ws = (char*)d_ws;
    _Float16* w1p  = (_Float16*)(ws + 0);                 // 2 KB
    _Float16* pw1p = (_Float16*)(ws + (4ull << 10));      // 2 KB
    _Float16* pw2p = (_Float16*)(ws + (8ull << 10));      // 4 KB
    _Float16* pw3p = (_Float16*)(ws + (16ull << 10));     // 16 KB
    float*    h0   = (float*)   (ws + (1ull << 20));      // 36.6 MB  [64,47,95,32]
    _Float16* s1   = (_Float16*)(ws + (40ull << 20));     // 4.4 MB   [69184,32]
    float*    p1   = (float*)   (ws + (45ull << 20));     // 8.9 MB   [69184,32]
    _Float16* s2   = (_Float16*)(ws + (54ull << 20));     // 1.0 MB   [16192,32]
    float*    p2   = (float*)   (ws + (56ull << 20));     // 4.1 MB   [16192,64]
    _Float16* s3   = (_Float16*)(ws + (61ull << 20));     // 0.45 MB  [3520,64]

    // weight packing (sign + WMMA B-fragment layout)
    pack_signB<<<1,  64, 0, stream>>>(w1,     w1p,  27,  32, 1, 2);
    pack_signB<<<1,  64, 0, stream>>>(pwk[0], pw1p, 32,  32, 1, 2);
    pack_signB<<<1, 128, 0, stream>>>(pwk[1], pw2p, 32,  64, 1, 4);
    pack_signB<<<2, 256, 0, stream>>>(pwk[2], pw3p, 64, 128, 2, 8);

    // stage 0: conv1 + pool + bn1  -> h0 [64,47,95,32]
    conv1_pool_bn<<<dim3(6, 47, 64), 32, 0, stream>>>(x, w1p, bn1_bt, bn1_mn, bn1_vr, h0);

    // block 1: h0[64,47,95,32] -> s1[69184,32] -> p1[69184,32]
    {
        int tot = BATCH * 23 * 47 * 32;
        dw_pool_bn_sign<<<(tot + 255) / 256, 256, 0, stream>>>(
            h0, dwk[0], dwb[0], Abt[0], Amn[0], Avr[0], s1, 47, 95, 32);
        pw_gemm_bn<1><<<dim3(69184 / 16, 2), 32, 0, stream>>>(
            s1, pw1p, Bbt[0], Bmn[0], Bvr[0], p1, 32, 2);
    }
    // block 2: p1[64,23,47,32] -> s2[16192,32] -> p2[16192,64]
    {
        int tot = BATCH * 11 * 23 * 32;
        dw_pool_bn_sign<<<(tot + 255) / 256, 256, 0, stream>>>(
            p1, dwk[1], dwb[1], Abt[1], Amn[1], Avr[1], s2, 23, 47, 32);
        pw_gemm_bn<1><<<dim3(16192 / 16, 4), 32, 0, stream>>>(
            s2, pw2p, Bbt[1], Bmn[1], Bvr[1], p2, 64, 4);
    }
    // block 3: p2[64,11,23,64] -> s3[3520,64] -> d_out[3520,128]
    {
        int tot = BATCH * 5 * 11 * 64;
        dw_pool_bn_sign<<<(tot + 255) / 256, 256, 0, stream>>>(
            p2, dwk[2], dwb[2], Abt[2], Amn[2], Avr[2], s3, 11, 23, 64);
        pw_gemm_bn<2><<<dim3(3520 / 16, 8), 32, 0, stream>>>(
            s3, pw3p, Bbt[2], Bmn[2], Bvr[2], outp, 128, 8);
    }
}
